// Decoder_7249904796235
// MI455X (gfx1250) — matrix operand order repaired and checked by reading, not compiled, not run on hardware
//
#include <hip/hip_runtime.h>
#include <hip/hip_bf16.h>

// ---------------- problem constants ----------------
#define V_    10004
#define PAD_  10003
#define VPAD  10240   // vocab padded to multiple of 128 (zero-filled)
#define E_    512
#define H_    512
#define B_    64
#define T_    64

// ---------------- WMMA types ----------------
typedef __attribute__((ext_vector_type(8)))  __bf16 v8bf;
typedef __attribute__((ext_vector_type(16))) __bf16 v16bf;
typedef __attribute__((ext_vector_type(8)))  float  v8f;

typedef unsigned int u32x4 __attribute__((ext_vector_type(4)));
typedef int          i32x4 __attribute__((ext_vector_type(4)));
typedef int          i32x8 __attribute__((ext_vector_type(8)));

#if __has_builtin(__builtin_amdgcn_tensor_load_to_lds)
#define HAVE_TDM 1
#endif

union Frag { v16bf v; v8bf h[2]; };

// A-matrix (16x32 bf16) fragment: row M = mbase + (lane&15); per-lane two
// contiguous 16B chunks at K offsets half*8 and 16+half*8 (ISA 7.12.2).
__device__ __forceinline__ v16bf load_A_frag(const __bf16* __restrict__ base,
                                             int stride, int mbase, int kb, int lane) {
  int m    = mbase + (lane & 15);
  int half = lane >> 4;
  const __bf16* p = base + (size_t)m * stride + kb + half * 8;
  Frag f;
  f.h[0] = *(const v8bf*)(p);
  f.h[1] = *(const v8bf*)(p + 16);
  return f.v;
}

// B-matrix (32x16 bf16) fragment from row-major W^T [K][N]: lane holds
// column N = nb + (lane&15); K offsets half*8+s and 16+half*8+s (s=0..7), same as A.
__device__ __forceinline__ v16bf load_B_frag(const __bf16* __restrict__ base,
                                             int stride, int kb, int nb, int lane) {
  int k = kb + (lane >> 4) * 8;
  const __bf16* p = base + (size_t)k * stride + nb + (lane & 15);
  Frag f;
  #pragma unroll
  for (int s = 0; s < 8; s++) {
    f.v[s]     = p[(size_t)s * stride];
    f.v[s + 8] = p[(size_t)(s + 16) * stride];
  }
  return f.v;
}

__device__ __forceinline__ v8f wmma_bf16(v16bf a, v16bf b, v8f c) {
  // (neg_a, A, neg_b, B, c_mod, C, reuse_a, reuse_b)
  return __builtin_amdgcn_wmma_f32_16x16x32_bf16(false, a, false, b, (short)0, c, false, false);
}

__device__ __forceinline__ float sigmoidf_(float x) { return 1.0f / (1.0f + expf(-x)); }

// ---------------- TDM: 64x128 bf16 2D tile -> LDS with hw row padding --------
// D# describes: data_size=2B, tile 128(x) x 64(y), tensor_dim0_stride=512,
// LDS pad 4 DWORDs after every 64 DWORDs -> LDS row stride 136 bf16 elements
// (matches the conflict-free As layout below).
__device__ __forceinline__ void tdm_load_tile_64x128(unsigned int lds_off,
                                                     const __bf16* gptr) {
#if defined(HAVE_TDM) && defined(__AMDGCN__)
  unsigned long long ga = (unsigned long long)(uintptr_t)gptr;
  u32x4 g0;
  g0.x = 1u;                                                 // count=1, user D#
  g0.y = lds_off;                                            // lds_addr (bytes)
  g0.z = (unsigned int)ga;                                   // global_addr[31:0]
  g0.w = (unsigned int)((ga >> 32) & 0x01FFFFFFu) | (2u << 30); // addr[56:32]|type=2
  i32x8 g1;
  g1[0] = (int)((1u << 16) |          // data_size = 2 bytes
                (1u << 20) |          // pad_enable
                (5u << 22) |          // pad_interval: 64 DWORDs
                (3u << 25));          // pad_amount:   4 DWORDs
  g1[1] = (int)(512u << 16);          // tensor_dim0 = 512 (bits 79:48 low half)
  g1[2] = (int)(64u << 16);           // tensor_dim0 hi=0 | tensor_dim1 = 64
  g1[3] = (int)(128u << 16);          // tensor_dim1 hi=0 | tile_dim0 = 128
  g1[4] = (int)64;                    // tile_dim1 = 64 | tile_dim2 = 0
  g1[5] = (int)512;                   // tensor_dim0_stride[31:0] = 512
  g1[6] = 0;                          // stride hi | tensor_dim1_stride lo
  g1[7] = 0;
  i32x4 z4 = {0, 0, 0, 0};
#if __clang_major__ >= 23
  i32x8 z8 = {0, 0, 0, 0, 0, 0, 0, 0};
  __builtin_amdgcn_tensor_load_to_lds(g0, g1, z4, z4, z8, 0);
#else
  __builtin_amdgcn_tensor_load_to_lds(g0, g1, z4, z4, 0);
#endif
#else
  (void)lds_off; (void)gptr;
#endif
}

// ---------------- prep: tiled transpose + fp32->bf16 cast ----------------
// dst[c*Rpad + r] = (r<R && r!=maskRow) ? src[r*C + c] : 0
__global__ __launch_bounds__(256) void transpose_cast(const float* __restrict__ src,
                                                      __bf16* __restrict__ dst,
                                                      int R, int C, int Rpad, int maskRow) {
  __shared__ float tile[32][33];
  int rt = blockIdx.y * 32, ct = blockIdx.x * 32;
  int tx = threadIdx.x, ty = threadIdx.y;   // 32 x 8
  #pragma unroll
  for (int i = 0; i < 32; i += 8) {
    int r = rt + ty + i, c = ct + tx;
    float v = 0.0f;
    if (r < R && c < C && r != maskRow) v = src[(size_t)r * C + c];
    tile[ty + i][tx] = v;
  }
  __syncthreads();
  #pragma unroll
  for (int i = 0; i < 32; i += 8) {
    int c = ct + ty + i, r = rt + tx;
    if (c < C && r < Rpad) dst[(size_t)c * Rpad + r] = (__bf16)tile[tx][ty + i];
  }
}

// ---------------- prep: gather all embeddings (masked) to bf16 ----------------
__global__ __launch_bounds__(512) void gather_emb(const int* __restrict__ x,
                                                  const float* __restrict__ emb,
                                                  __bf16* __restrict__ e_all) {
  int tb = blockIdx.x;                 // tb = t*64 + b
  int t = tb >> 6, b = tb & 63;
  int tok = x[b * T_ + t];             // x is [B][T]
  int k = threadIdx.x;                 // 512 threads = E
  float v = (tok == PAD_) ? 0.0f : emb[(size_t)tok * E_ + k];
  e_all[(size_t)tb * E_ + k] = (__bf16)v;
}

// ---------------- prep: ses = tanh(lin1(ses_enc)), init recurrent state ----------------
__global__ __launch_bounds__(512) void ses_init(const float* __restrict__ senc,
                                                const float* __restrict__ W,   // [512][1024]
                                                const float* __restrict__ bias,
                                                float* __restrict__ ses_f32,
                                                float* __restrict__ hrnn_f0,
                                                __bf16* __restrict__ hfull0,
                                                __bf16* __restrict__ hfull1,
                                                float* __restrict__ hlm_f0,
                                                __bf16* __restrict__ hlm_b0) {
  int b = blockIdx.x;      // 64
  int j = threadIdx.x;     // 512
  const float* s = senc + (size_t)b * 1024;
  const float* w = W + (size_t)j * 1024;
  float acc = bias[j];
  for (int k = 0; k < 1024; k++) acc = fmaf(s[k], w[k], acc);
  float v = tanhf(acc);
  __bf16 vb = (__bf16)v;
  ses_f32[b * H_ + j] = v;
  hrnn_f0[b * H_ + j] = v;                 // h0 = ses
  hfull0[b * 1024 + j]       = vb;         // concat(h_rnn, ses)
  hfull0[b * 1024 + 512 + j] = vb;         // ses half (buffer 0)
  hfull1[b * 1024 + 512 + j] = vb;         // ses half (buffer 1, never rewritten)
  hlm_f0[b * H_ + j] = 0.0f;
  hlm_b0[b * H_ + j] = (__bf16)0.0f;
}

// ---------------- GRU step (one cell, WMMA) ----------------
// gi = e @ WihT (+bih), gh = h @ WhhT (+bhh); gates (r,z,n); h' = (1-z)*n + z*h
__global__ __launch_bounds__(128) void gru_step(
    const __bf16* __restrict__ Ae,        // [64][Ke]      input embeddings (this t)
    const __bf16* __restrict__ Ah,        // [64][Kh]      hidden state bf16
    const __bf16* __restrict__ WihT,      // [Ke][3*Nh]
    const __bf16* __restrict__ WhhT,      // [Kh][3*Nh]
    const float*  __restrict__ bih,
    const float*  __restrict__ bhh,
    const float*  __restrict__ hstate_in, // [64][SC] fp32 recurrent state
    const float*  __restrict__ ses_f32,   // [64][512] (cols >= SC come from ses)
    __bf16* __restrict__ hnew_bf,         // [64][Nh]  full new hidden (for projection)
    float*  __restrict__ hstate_out,      // [64][SC]
    __bf16* __restrict__ hstate_bf_out,   // stride SOstride, first SC cols
    int Ke, int Kh, int Nh, int SC, int SOstride) {
  int lane = threadIdx.x & 31;
  int wave = threadIdx.x >> 5;
  int mbase = wave * 16;                  // 4 waves cover B=64
  int jb = blockIdx.x * 16;               // hidden-column tile

  v8f gir = {}, giz = {}, gin = {}, ghr = {}, ghz = {}, ghn = {};
  int WiS = 3 * Nh, WhS = 3 * Nh;

  for (int kb = 0; kb < Ke; kb += 32) {
    v16bf a  = load_A_frag(Ae, Ke, mbase, kb, lane);
    v16bf b0 = load_B_frag(WihT, WiS, kb, 0 * Nh + jb, lane);
    v16bf b1 = load_B_frag(WihT, WiS, kb, 1 * Nh + jb, lane);
    v16bf b2 = load_B_frag(WihT, WiS, kb, 2 * Nh + jb, lane);
    gir = wmma_bf16(a, b0, gir);
    giz = wmma_bf16(a, b1, giz);
    gin = wmma_bf16(a, b2, gin);
  }
  for (int kb = 0; kb < Kh; kb += 32) {
    v16bf a  = load_A_frag(Ah, Kh, mbase, kb, lane);
    v16bf b0 = load_B_frag(WhhT, WhS, kb, 0 * Nh + jb, lane);
    v16bf b1 = load_B_frag(WhhT, WhS, kb, 1 * Nh + jb, lane);
    v16bf b2 = load_B_frag(WhhT, WhS, kb, 2 * Nh + jb, lane);
    ghr = wmma_bf16(a, b0, ghr);
    ghz = wmma_bf16(a, b1, ghz);
    ghn = wmma_bf16(a, b2, ghn);
  }

  int n = lane & 15, halfM = lane >> 4;
  int j = jb + n;
  float br = bih[j],        bz = bih[Nh + j],      bn = bih[2 * Nh + j];
  float cr = bhh[j],        cz = bhh[Nh + j],      cn = bhh[2 * Nh + j];
  #pragma unroll
  for (int i = 0; i < 8; i++) {
    int row = mbase + i + 8 * halfM;    // batch index
    float r  = sigmoidf_(gir[i] + ghr[i] + br + cr);
    float z  = sigmoidf_(giz[i] + ghz[i] + bz + cz);
    float nn = tanhf(gin[i] + bn + r * (ghn[i] + cn));
    float hp = (j < SC) ? hstate_in[row * SC + j] : ses_f32[row * 512 + (j - SC)];
    float h  = (1.0f - z) * nn + z * hp;
    hnew_bf[(size_t)row * Nh + j] = (__bf16)h;
    if (j < SC) {
      hstate_out[row * SC + j] = h;
      hstate_bf_out[(size_t)row * SOstride + j] = (__bf16)h;
    }
  }
}

// ---------------- per-step projection: P[b*T+t][0:512] = h @ linT ----------------
__global__ __launch_bounds__(128) void proj_gemm(const __bf16* __restrict__ A,   // [64][K]
                                                 const __bf16* __restrict__ WT,  // [K][512]
                                                 __bf16* __restrict__ P,         // [4096][512]
                                                 int K, int t) {
  int lane = threadIdx.x & 31;
  int wave = threadIdx.x >> 5;
  int mbase = wave * 16;
  int nb = blockIdx.x * 16;               // 32 blocks cover N=512
  v8f acc = {};
  for (int kb = 0; kb < K; kb += 32) {
    v16bf a = load_A_frag(A, K, mbase, kb, lane);
    v16bf b = load_B_frag(WT, 512, kb, nb, lane);
    acc = wmma_bf16(a, b, acc);
  }
  int n = lane & 15, halfM = lane >> 4;
  #pragma unroll
  for (int i = 0; i < 8; i++) {
    int bRow = mbase + i + 8 * halfM;     // batch
    int r = bRow * T_ + t;                // output row ordering = (b*T + t)
    P[(size_t)r * 512 + nb + n] = (__bf16)acc[i];
  }
}

// ---------------- bulk logits GEMM: out[o][r][v] = P[o][r][:] . embT[:][v] ----------------
// block: 256 thr = 8 waves (4 m-tiles x 2 n-halves); block tile 64(M) x 128(N); K=512.
// A tile double-buffered in LDS, staged by the Tensor Data Mover (TDM) with
// hardware row padding (136-element stride, conflict-free ds_load_b128).
__global__ __launch_bounds__(256) void logits_gemm(const __bf16* __restrict__ Pall, // [2][4096][512]
                                                   const __bf16* __restrict__ embT, // [512][VPAD]
                                                   float* __restrict__ out) {
  __shared__ __bf16 As[2][64 * 136];
  int o = blockIdx.z;
  const __bf16* Ap = Pall + (size_t)o * 4096 * 512;
  float* Op = out + (size_t)o * 4096 * V_;
  int rbase = blockIdx.y * 64;
  int nb0 = blockIdx.x * 128;
  int tid = threadIdx.x;
  int lane = tid & 31, wave = tid >> 5;
  int wm = wave & 3, wh = wave >> 2;

#if defined(HAVE_TDM)
  if (wave == 0) {
    tdm_load_tile_64x128((unsigned int)(uintptr_t)&As[0][0],
                         Ap + (size_t)rbase * 512);
  }
#endif

  v8f acc[4] = {};
  for (int kc = 0; kc < 4; kc++) {        // 4 K-chunks of 128
    int buf = kc & 1;
#if defined(HAVE_TDM)
    if (wave == 0) __builtin_amdgcn_s_wait_tensorcnt(0);  // chunk kc landed in LDS
    __syncthreads();                                      // publish to all waves
    if (wave == 0 && kc < 3) {                            // overlap next DMA w/ compute
      tdm_load_tile_64x128((unsigned int)(uintptr_t)&As[buf ^ 1][0],
                           Ap + (size_t)rbase * 512 + (kc + 1) * 128);
    }
#else
    __syncthreads();
    // cooperative stage of A chunk: 64x128 bf16 (16B vector per thread-iter)
    for (int c = tid; c < 1024; c += 256) {
      int row = c >> 4, cc = (c & 15) * 8;
      v8bf vv = *(const v8bf*)(Ap + (size_t)(rbase + row) * 512 + kc * 128 + cc);
      *(v8bf*)(&As[buf][row * 136 + cc]) = vv;
    }
    __syncthreads();
    if (kc < 3)  // warm next A chunk into caches
      __builtin_prefetch(Ap + (size_t)(rbase + (tid >> 2)) * 512 + (kc + 1) * 128 + (tid & 3) * 32, 0, 1);
#endif

    for (int k32 = 0; k32 < 4; k32++) {
      int kb = k32 * 32;
      v16bf a;
      {
        int m = wm * 16 + (lane & 15);
        int half = lane >> 4;
        const __bf16* p = &As[buf][m * 136 + kb + half * 8];
        Frag f;
        f.h[0] = *(const v8bf*)(p);
        f.h[1] = *(const v8bf*)(p + 16);
        a = f.v;
      }
      int kglob = kc * 128 + kb;
      #pragma unroll
      for (int aI = 0; aI < 4; aI++) {
        v16bf b = load_B_frag(embT, VPAD, kglob, nb0 + wh * 64 + aI * 16, lane);
        acc[aI] = wmma_bf16(a, b, acc[aI]);
      }
    }

#if defined(HAVE_TDM)
    __syncthreads();   // all reads of As[buf] done before TDM reuses it
#endif
  }

  int n = lane & 15, halfM = lane >> 4;
  #pragma unroll
  for (int aI = 0; aI < 4; aI++) {
    int v = nb0 + wh * 64 + aI * 16 + n;
    if (v < V_) {
      #pragma unroll
      for (int i = 0; i < 8; i++) {
        int r = rbase + wm * 16 + i + 8 * halfM;   // r = b*T + t
        Op[(size_t)r * V_ + v] = acc[aI][i];
      }
    }
  }
}

// ---------------- host launcher ----------------
extern "C" void kernel_launch(void* const* d_in, const int* in_sizes, int n_in,
                              void* d_out, int out_size, void* d_ws, size_t ws_size,
                              hipStream_t stream) {
  (void)in_sizes; (void)n_in; (void)out_size; (void)ws_size;
  const float* ses_enc = (const float*)d_in[0];
  const int*   x       = (const int*)d_in[1];
  const float* emb     = (const float*)d_in[2];
  const float* rnn_Wih = (const float*)d_in[3];
  const float* rnn_Whh = (const float*)d_in[4];
  const float* rnn_bih = (const float*)d_in[5];
  const float* rnn_bhh = (const float*)d_in[6];
  const float* lm_Wih  = (const float*)d_in[7];
  const float* lm_Whh  = (const float*)d_in[8];
  const float* lm_bih  = (const float*)d_in[9];
  const float* lm_bhh  = (const float*)d_in[10];
  const float* lin1_W  = (const float*)d_in[11];
  const float* lin1_b  = (const float*)d_in[12];
  const float* lin2_W  = (const float*)d_in[13];
  const float* lin3_W  = (const float*)d_in[14];
  float* out = (float*)d_out;

  char* wsb = (char*)d_ws;
  size_t off = 0;
  auto alloc = [&](size_t bytes) -> void* {
    void* p = wsb + off;
    off += (bytes + 255) & ~(size_t)255;
    return p;
  };
  __bf16* embT     = (__bf16*)alloc((size_t)E_ * VPAD * 2);
  __bf16* e_all    = (__bf16*)alloc((size_t)T_ * B_ * E_ * 2);
  __bf16* WihT_rnn = (__bf16*)alloc((size_t)512 * 3072 * 2);
  __bf16* WhhT_rnn = (__bf16*)alloc((size_t)1024 * 3072 * 2);
  __bf16* WihT_lm  = (__bf16*)alloc((size_t)512 * 1536 * 2);
  __bf16* WhhT_lm  = (__bf16*)alloc((size_t)512 * 1536 * 2);
  __bf16* lin2T    = (__bf16*)alloc((size_t)1024 * 512 * 2);
  __bf16* lin3T    = (__bf16*)alloc((size_t)512 * 512 * 2);
  float*  ses_f    = (float*)alloc((size_t)B_ * H_ * 4);
  float*  hrnn_f[2] = {(float*)alloc((size_t)B_ * H_ * 4), (float*)alloc((size_t)B_ * H_ * 4)};
  __bf16* hfull[2]  = {(__bf16*)alloc((size_t)B_ * 1024 * 2), (__bf16*)alloc((size_t)B_ * 1024 * 2)};
  float*  hlm_f[2]  = {(float*)alloc((size_t)B_ * H_ * 4), (float*)alloc((size_t)B_ * H_ * 4)};
  __bf16* hlm_b[2]  = {(__bf16*)alloc((size_t)B_ * H_ * 2), (__bf16*)alloc((size_t)B_ * H_ * 2)};
  __bf16* hnew      = (__bf16*)alloc((size_t)B_ * 1024 * 2);
  __bf16* hnewlm    = (__bf16*)alloc((size_t)B_ * H_ * 2);
  __bf16* P         = (__bf16*)alloc((size_t)2 * 4096 * 512 * 2);

  dim3 tpb(32, 8);
  // --- prep: transpose+cast to bf16 W^T layouts ---
  transpose_cast<<<dim3(E_ / 32, VPAD / 32), tpb, 0, stream>>>(emb, embT, V_, E_, VPAD, PAD_);
  transpose_cast<<<dim3(512 / 32, 3072 / 32), tpb, 0, stream>>>(rnn_Wih, WihT_rnn, 3072, 512, 3072, -1);
  transpose_cast<<<dim3(1024 / 32, 3072 / 32), tpb, 0, stream>>>(rnn_Whh, WhhT_rnn, 3072, 1024, 3072, -1);
  transpose_cast<<<dim3(512 / 32, 1536 / 32), tpb, 0, stream>>>(lm_Wih, WihT_lm, 1536, 512, 1536, -1);
  transpose_cast<<<dim3(512 / 32, 1536 / 32), tpb, 0, stream>>>(lm_Whh, WhhT_lm, 1536, 512, 1536, -1);
  transpose_cast<<<dim3(1024 / 32, 512 / 32), tpb, 0, stream>>>(lin2_W, lin2T, 512, 1024, 512, -1);
  transpose_cast<<<dim3(512 / 32, 512 / 32), tpb, 0, stream>>>(lin3_W, lin3T, 512, 512, 512, -1);
  gather_emb<<<T_ * B_, 512, 0, stream>>>(x, emb, e_all);
  ses_init<<<B_, 512, 0, stream>>>(ses_enc, lin1_W, lin1_b, ses_f,
                                   hrnn_f[0], hfull[0], hfull[1], hlm_f[0], hlm_b[0]);

  // --- sequential GRU scan ---
  for (int t = 0; t < T_; t++) {
    int rb = t & 1, wb = rb ^ 1;
    const __bf16* Ae = e_all + (size_t)t * B_ * E_;
    // main GRU: Nh = 2H = 1024, hidden input = concat(h_rnn, ses)
    gru_step<<<64, 128, 0, stream>>>(Ae, hfull[rb], WihT_rnn, WhhT_rnn, rnn_bih, rnn_bhh,
                                     hrnn_f[rb], ses_f, hnew, hrnn_f[wb], hfull[wb],
                                     /*Ke=*/512, /*Kh=*/1024, /*Nh=*/1024, /*SC=*/512, /*SOstride=*/1024);
    // lm GRU: Nh = H = 512
    gru_step<<<32, 128, 0, stream>>>(Ae, hlm_b[rb], WihT_lm, WhhT_lm, lm_bih, lm_bhh,
                                     hlm_f[rb], ses_f, hnewlm, hlm_f[wb], hlm_b[wb],
                                     /*Ke=*/512, /*Kh=*/512, /*Nh=*/512, /*SC=*/512, /*SOstride=*/512);
    // projections into P rows (b*T + t)
    proj_gemm<<<32, 128, 0, stream>>>(hnew, lin2T, P, /*K=*/1024, t);
    proj_gemm<<<32, 128, 0, stream>>>(hnewlm, lin3T, P + (size_t)4096 * 512, /*K=*/512, t);
  }

  // --- bulk tied-embedding logits GEMM (both outputs) ---
  logits_gemm<<<dim3(VPAD / 128, (B_ * T_) / 64, 2), 256, 0, stream>>>(P, embT, out);
}
